// DeepSpeedMoeWithJitter_3126736191797
// MI455X (gfx1250) — matrix operand order repaired and checked by reading, not compiled, not run on hardware
//
#include <hip/hip_runtime.h>
#include <math.h>
#include <stdint.h>

// ---------------- types ----------------
typedef __bf16 bf16_t;
typedef bf16_t v16bf __attribute__((ext_vector_type(16)));
typedef bf16_t v8bf  __attribute__((ext_vector_type(8)));
typedef float  v8f   __attribute__((ext_vector_type(8)));

union V16 { v16bf v; v8bf h[2]; };

// ---------------- problem constants ----------------
#define BATCH   8192
#define IN_DIM  4096
#define HID     2048
#define NEXP    6
#define NC      1000
#define NCPAD   1024
#define CAPSZ   2731      // ceil(8192*2/6)
#define CAPP    2816      // CAP padded to multiple of 128 for tiling
#define GT_BLK  8         // tokens per gating block (8 waves of 32)
#define NGBLK   (BATCH / GT_BLK)   // 1024

// ---------------- CDNA5 async helpers ----------------
// GLOBAL_LOAD_ASYNC_TO_LDS_B128: DMA 16B/lane global->LDS, tracked on ASYNCcnt.
// INST_OFFSET is added to BOTH the LDS and global address (ISA 08 §4.4), so one
// base pair + offset:16 covers each lane's second 16B chunk.
__device__ __forceinline__ void async_copy_32B(unsigned lds_addr, const void* gaddr)
{
    asm volatile("global_load_async_to_lds_b128 %0, %1, off"
                 :: "v"(lds_addr), "v"(gaddr) : "memory");
    asm volatile("global_load_async_to_lds_b128 %0, %1, off offset:16"
                 :: "v"(lds_addr), "v"(gaddr) : "memory");
}

__device__ __forceinline__ void wait_async0()
{
#if __has_builtin(__builtin_amdgcn_s_wait_asynccnt)
    __builtin_amdgcn_s_wait_asynccnt(0);
#else
    asm volatile("s_wait_asynccnt 0x0" ::: "memory");
#endif
}

// flat shared pointer -> LDS byte address (low 32 bits of the flat address)
__device__ __forceinline__ unsigned lds_off(const void* p)
{
    return (unsigned)(uintptr_t)p;
}

// =====================================================================
// Tiled bf16 WMMA GEMM:  C[M,N] = act(A[M,K] * B^T[N,K] + bias)
// A: f32 (converted in-flight) or bf16 (async-DMA'd).  B: bf16 [N][K],
// async-DMA'd global->LDS. Double-buffered LDS (48KB), one barrier/K-step.
// Block tile 128x128x32, 256 threads = 8 waves; wave tile 32x64
// (2x4 grid of 16x16 frags), v_wmma_f32_16x16x32_bf16, f32 accumulate.
// =====================================================================
template<bool AF32, bool CF32, bool RELU, bool BIAS>
__global__ __launch_bounds__(256)
void gemm_wmma_bf16(const void* __restrict__ Ap, const bf16_t* __restrict__ BTp,
                    void* __restrict__ Cp, const float* __restrict__ bias,
                    int M, int N, int K,
                    long long aBatch, long long bBatch, long long cBatch,
                    long long biasBatch)
{
    const long long ez = blockIdx.z;

    const float*  Af = nullptr;
    const bf16_t* Ab = nullptr;
    if constexpr (AF32) Af = (const float*)Ap  + ez * aBatch;
    else                Ab = (const bf16_t*)Ap + ez * aBatch;
    const bf16_t* BT = BTp + ez * bBatch;

    const int m0 = blockIdx.y * 128;
    const int n0 = blockIdx.x * 128;

    // +16 bf16 pad keeps every 16B fragment load 16B-aligned and skews banks
    __shared__ bf16_t As[2][128][48];
    __shared__ bf16_t Bs[2][128][48];

    const int tid  = threadIdx.x;
    const int w    = tid >> 5;        // wave id 0..7
    const int lane = tid & 31;
    const int wm   = w & 3;           // 4 waves along M
    const int wn   = w >> 2;          // 2 waves along N
    const int lr   = lane & 15;       // lane row/col within frag
    const int lh   = lane >> 4;       // lane half (K-half selector)

    v8f acc[2][4] = {};               // 8 f32 accumulators (64 VGPRs)

    // cooperative tile-load mapping: 256 threads, 128 rows x 32 cols bf16
    const int trow = tid >> 1;        // 0..127
    const int tc0  = (tid & 1) * 16;  // 0 or 16

    const int KT = K >> 5;

    // ---- tile staging (into buffer q) ----
    auto stageA = [&](int kt, int q) {
        const int k0 = kt << 5;
        if constexpr (AF32) {
            // f32 -> bf16 conversion must go through VGPRs
            const float* src = Af + (long long)(m0 + trow) * K + k0 + tc0;
            if (kt + 1 < KT) __builtin_prefetch(src + 32, 0, 0);  // global_prefetch_b8
#pragma unroll
            for (int j = 0; j < 4; ++j) {
                float4 f = *(const float4*)(src + 4 * j);
                bf16_t* d = &As[q][trow][tc0 + 4 * j];
                d[0] = (bf16_t)f.x; d[1] = (bf16_t)f.y;
                d[2] = (bf16_t)f.z; d[3] = (bf16_t)f.w;
            }
        } else {
            // pure byte copy: async DMA, no VGPR round-trip
            const bf16_t* src = Ab + (long long)(m0 + trow) * K + k0 + tc0;
            async_copy_32B(lds_off(&As[q][trow][tc0]), src);
        }
    };
    auto stageB = [&](int kt, int q) {
        const bf16_t* src = BT + (long long)(n0 + trow) * K + (kt << 5) + tc0;
        async_copy_32B(lds_off(&Bs[q][trow][tc0]), src);
    };

    // ---- prologue: stage tile 0 into buffer 0 ----
    stageA(0, 0);
    stageB(0, 0);
    wait_async0();
    __syncthreads();

    for (int kt = 0; kt < KT; ++kt) {
        const int p = kt & 1;
        // stream next tile into the other buffer while computing this one
        if (kt + 1 < KT) {
            stageA(kt + 1, p ^ 1);
            stageB(kt + 1, p ^ 1);
        }

        // ---- fragments per WMMA lane layout ----
        // A 16x32: lanes 0-15 rows M, K 0..7 & 16..23; lanes 16-31 K 8..15 & 24..31
        V16 afr[2];
#pragma unroll
        for (int mf = 0; mf < 2; ++mf) {
            const int r = wm * 32 + mf * 16 + lr;
            afr[mf].h[0] = *(const v8bf*)&As[p][r][lh * 8];
            afr[mf].h[1] = *(const v8bf*)&As[p][r][16 + lh * 8];
        }
        // B 32x16: lanes 0-15 cols N, K 0..7 & 16..23; lanes 16-31 K 8..15 & 24..31
        V16 bfr[4];
#pragma unroll
        for (int nf = 0; nf < 4; ++nf) {
            const int c = wn * 64 + nf * 16 + lr;
            bfr[nf].h[0] = *(const v8bf*)&Bs[p][c][lh * 8];
            bfr[nf].h[1] = *(const v8bf*)&Bs[p][c][16 + lh * 8];
        }

#pragma unroll
        for (int mf = 0; mf < 2; ++mf)
#pragma unroll
            for (int nf = 0; nf < 4; ++nf)
                acc[mf][nf] = __builtin_amdgcn_wmma_f32_16x16x32_bf16(
                    false, afr[mf].v, false, bfr[nf].v,
                    (short)0, acc[mf][nf], false, false);

        // my async writes into buffer p^1 must land before anyone reads them;
        // frag ds_loads of buffer p were consumed by the WMMAs above
        wait_async0();
        __syncthreads();
    }

    // ---- epilogue: bias + relu + store (f32 or bf16) ----
    float*  Cf = nullptr; bf16_t* Cb = nullptr;
    if constexpr (CF32) Cf = (float*)Cp  + ez * cBatch;
    else                Cb = (bf16_t*)Cp + ez * cBatch;

#pragma unroll
    for (int mf = 0; mf < 2; ++mf) {
#pragma unroll
        for (int nf = 0; nf < 4; ++nf) {
            const int colg = n0 + wn * 64 + nf * 16 + lr;
            float bv = 0.f;
            if constexpr (BIAS) bv = bias[ez * biasBatch + colg];
#pragma unroll
            for (int r = 0; r < 8; ++r) {
                const int rowg = m0 + wm * 32 + mf * 16 + lh * 8 + r;
                float v = acc[mf][nf][r] + bv;
                if constexpr (RELU) v = fmaxf(v, 0.f);
                const long long ci = (long long)rowg * N + colg;
                if constexpr (CF32) Cf[ci] = v;
                else                Cb[ci] = (bf16_t)v;
            }
        }
    }
}

// =====================================================================
// Transpose + f32->bf16 cast:  WT[n][k] = (bf16) W[k][n], n padded/zeroed
// =====================================================================
__global__ __launch_bounds__(256)
void transpose_cast_kernel(const float* __restrict__ W, bf16_t* __restrict__ WT,
                           int K, int N, long long wBatch, long long wtBatch)
{
    __shared__ float tile[32][33];
    const float* src = W  + (long long)blockIdx.z * wBatch;
    bf16_t*      dst = WT + (long long)blockIdx.z * wtBatch;
    const int k0 = blockIdx.x * 32, n0 = blockIdx.y * 32;
    const int tx = threadIdx.x, ty = threadIdx.y;   // 32 x 8
#pragma unroll
    for (int j = 0; j < 4; ++j) {
        const int k = k0 + ty + j * 8, n = n0 + tx;
        tile[ty + j * 8][tx] = (n < N) ? src[(long long)k * N + n] : 0.f;
    }
    __syncthreads();
#pragma unroll
    for (int j = 0; j < 4; ++j) {
        const int n = n0 + ty + j * 8, k = k0 + tx;
        dst[(long long)n * K + k] = (bf16_t)tile[tx][ty + j * 8];
    }
}

// =====================================================================
// Gating: logits = h @ Wg, softmax, top-1/top-2 + per-block histograms
// one wave (32 lanes) per token, 8 tokens per block
// =====================================================================
__global__ __launch_bounds__(256)
void gate_kernel(const bf16_t* __restrict__ h, const float* __restrict__ Wg,
                 int* __restrict__ idx1, int* __restrict__ idx2,
                 float* __restrict__ g1, float* __restrict__ g2,
                 int* __restrict__ hist1, int* __restrict__ hist2)
{
    __shared__ int s1[NEXP], s2[NEXP];
    const int tid = threadIdx.x;
    if (tid < NEXP) { s1[tid] = 0; s2[tid] = 0; }
    __syncthreads();

    const int wv = tid >> 5, lane = tid & 31;
    const int t  = blockIdx.x * GT_BLK + wv;

    float p[NEXP];
#pragma unroll
    for (int e = 0; e < NEXP; ++e) p[e] = 0.f;

    const bf16_t* hr = h + (long long)t * HID;
    for (int j = lane; j < HID; j += 32) {
        const float hv = (float)hr[j];
        const float* wr = Wg + j * NEXP;
#pragma unroll
        for (int e = 0; e < NEXP; ++e) p[e] += hv * wr[e];
    }
#pragma unroll
    for (int e = 0; e < NEXP; ++e)
        for (int m = 16; m >= 1; m >>= 1)
            p[e] += __shfl_xor(p[e], m, 32);

    if (lane == 0) {
        float mx = p[0];
#pragma unroll
        for (int e = 1; e < NEXP; ++e) mx = fmaxf(mx, p[e]);
        float gg[NEXP], sum = 0.f;
#pragma unroll
        for (int e = 0; e < NEXP; ++e) { gg[e] = expf(p[e] - mx); sum += gg[e]; }
#pragma unroll
        for (int e = 0; e < NEXP; ++e) gg[e] /= sum;
        int i1 = 0;
#pragma unroll
        for (int e = 1; e < NEXP; ++e) if (p[e] > p[i1]) i1 = e;
        int i2 = -1;
#pragma unroll
        for (int e = 0; e < NEXP; ++e)
            if (e != i1 && (i2 < 0 || p[e] > p[i2])) i2 = e;
        idx1[t] = i1; idx2[t] = i2;
        g1[t] = gg[i1]; g2[t] = gg[i2];
        atomicAdd(&s1[i1], 1); atomicAdd(&s2[i2], 1);
    }
    __syncthreads();
    if (tid < NEXP) {
        hist1[blockIdx.x * NEXP + tid] = s1[tid];
        hist2[blockIdx.x * NEXP + tid] = s2[tid];
    }
}

// Exclusive scan of per-block histograms + grand totals (tiny, 1 thread)
__global__ void scan_kernel(const int* __restrict__ hist1, const int* __restrict__ hist2,
                            int* __restrict__ off1, int* __restrict__ off2,
                            int* __restrict__ T1, int nblk)
{
    int r1[NEXP] = {}, r2[NEXP] = {};
    for (int b = 0; b < nblk; ++b)
        for (int e = 0; e < NEXP; ++e) {
            off1[b * NEXP + e] = r1[e]; r1[e] += hist1[b * NEXP + e];
            off2[b * NEXP + e] = r2[e]; r2[e] += hist2[b * NEXP + e];
        }
    for (int e = 0; e < NEXP; ++e) T1[e] = r1[e];
}

// Token-ordered slot assignment with capacity drop + gate renormalization
__global__ void assign_kernel(const int* __restrict__ idx1, const int* __restrict__ idx2,
                              const float* __restrict__ g1, const float* __restrict__ g2,
                              const int* __restrict__ off1, const int* __restrict__ off2,
                              const int* __restrict__ T1,
                              int* __restrict__ slot1, int* __restrict__ slot2,
                              float* __restrict__ w1, float* __restrict__ w2)
{
    const int b = blockIdx.x;
    int c1[NEXP], c2[NEXP];
    for (int e = 0; e < NEXP; ++e) { c1[e] = off1[b * NEXP + e]; c2[e] = off2[b * NEXP + e]; }
    for (int i = 0; i < GT_BLK; ++i) {
        const int t = b * GT_BLK + i;
        const int e1 = idx1[t]; const int p1 = c1[e1]++;
        const int e2 = idx2[t]; const int p2 = T1[e2] + (c2[e2]++);
        const bool k1 = p1 < CAPSZ, k2 = p2 < CAPSZ;
        const float a1 = k1 ? g1[t] : 0.f;
        const float a2 = k2 ? g2[t] : 0.f;
        const float den = fmaxf(a1 + a2, 1.1920929e-07f);
        w1[t] = a1 / den; w2[t] = a2 / den;
        slot1[t] = k1 ? (e1 * CAPP + p1) : -1;
        slot2[t] = k2 ? (e2 * CAPP + p2) : -1;
    }
}

// Scatter token rows into per-expert capacity buffers (bf16)
__global__ __launch_bounds__(256)
void dispatch_kernel(const bf16_t* __restrict__ h,
                     const int* __restrict__ slot1, const int* __restrict__ slot2,
                     bf16_t* __restrict__ buf)
{
    const int t = blockIdx.x;
    const int c = threadIdx.x * 8;
    const v8bf v = *(const v8bf*)(h + (long long)t * HID + c);
    const int s1 = slot1[t];
    if (s1 >= 0) *(v8bf*)(buf + (long long)s1 * HID + c) = v;
    const int s2 = slot2[t];
    if (s2 >= 0) *(v8bf*)(buf + (long long)s2 * HID + c) = v;
}

// Gather + gate-weighted combine of expert outputs -> bf16 tokens
__global__ __launch_bounds__(256)
void combine_kernel(const bf16_t* __restrict__ eo,
                    const int* __restrict__ slot1, const int* __restrict__ slot2,
                    const float* __restrict__ w1, const float* __restrict__ w2,
                    bf16_t* __restrict__ h3)
{
    const int t = blockIdx.x;
    const int c = threadIdx.x * 8;
    const float a1 = w1[t], a2 = w2[t];
    const int s1 = slot1[t], s2 = slot2[t];
    float r[8] = {};
    if (s1 >= 0) {
        const v8bf v = *(const v8bf*)(eo + (long long)s1 * HID + c);
#pragma unroll
        for (int j = 0; j < 8; ++j) r[j] += a1 * (float)v[j];
    }
    if (s2 >= 0) {
        const v8bf v = *(const v8bf*)(eo + (long long)s2 * HID + c);
#pragma unroll
        for (int j = 0; j < 8; ++j) r[j] += a2 * (float)v[j];
    }
    v8bf o;
#pragma unroll
    for (int j = 0; j < 8; ++j) o[j] = (bf16_t)r[j];
    *(v8bf*)(h3 + (long long)t * HID + c) = o;
}

// Row-wise log_softmax over 1000 classes (+ bias), f32
__global__ __launch_bounds__(256)
void logsoftmax_kernel(const float* __restrict__ logits, const float* __restrict__ bp,
                       float* __restrict__ out)
{
    __shared__ float red[256];
    const int t = blockIdx.x, tid = threadIdx.x;
    const float* z = logits + (long long)t * NCPAD;

    float zv[4];
    float mx = -INFINITY;
#pragma unroll
    for (int j = 0; j < 4; ++j) {
        const int c = tid + j * 256;
        zv[j] = (c < NC) ? (z[c] + bp[c]) : -INFINITY;
        mx = fmaxf(mx, zv[j]);
    }
    red[tid] = mx; __syncthreads();
    for (int s = 128; s > 0; s >>= 1) {
        if (tid < s) red[tid] = fmaxf(red[tid], red[tid + s]);
        __syncthreads();
    }
    mx = red[0]; __syncthreads();

    float sum = 0.f;
#pragma unroll
    for (int j = 0; j < 4; ++j) {
        const int c = tid + j * 256;
        if (c < NC) sum += expf(zv[j] - mx);
    }
    red[tid] = sum; __syncthreads();
    for (int s = 128; s > 0; s >>= 1) {
        if (tid < s) red[tid] += red[tid + s];
        __syncthreads();
    }
    const float lse = logf(red[0]);
#pragma unroll
    for (int j = 0; j < 4; ++j) {
        const int c = tid + j * 256;
        if (c < NC) out[(long long)t * NC + c] = zv[j] - mx - lse;
    }
}

// =====================================================================
extern "C" void kernel_launch(void* const* d_in, const int* in_sizes, int n_in,
                              void* d_out, int out_size, void* d_ws, size_t ws_size,
                              hipStream_t stream)
{
    (void)in_sizes; (void)n_in; (void)out_size; (void)ws_size;
    const float* x  = (const float*)d_in[0];
    const float* W0 = (const float*)d_in[1];
    const float* b0 = (const float*)d_in[2];
    const float* W1 = (const float*)d_in[3];
    const float* b1 = (const float*)d_in[4];
    const float* Wg = (const float*)d_in[5];
    const float* We = (const float*)d_in[6];
    const float* be = (const float*)d_in[7];
    const float* Wp = (const float*)d_in[8];
    const float* bp = (const float*)d_in[9];
    float* out = (float*)d_out;

    char* ws = (char*)d_ws;
    size_t off = 0;
    auto alloc = [&](size_t bytes) -> char* {
        char* p = ws + off; off += (bytes + 255) & ~(size_t)255; return p;
    };

    bf16_t* W0T  = (bf16_t*)alloc((size_t)HID * IN_DIM * 2);
    bf16_t* W1T  = (bf16_t*)alloc((size_t)HID * HID * 2);
    bf16_t* WeT  = (bf16_t*)alloc((size_t)NEXP * HID * HID * 2);
    bf16_t* WpT  = (bf16_t*)alloc((size_t)NCPAD * HID * 2);
    bf16_t* h1   = (bf16_t*)alloc((size_t)BATCH * HID * 2);   // reused as h3
    bf16_t* h2   = (bf16_t*)alloc((size_t)BATCH * HID * 2);
    bf16_t* dbuf = (bf16_t*)alloc((size_t)NEXP * CAPP * HID * 2);
    bf16_t* eo   = (bf16_t*)alloc((size_t)NEXP * CAPP * HID * 2);
    int*   idx1  = (int*)alloc(BATCH * 4);
    int*   idx2  = (int*)alloc(BATCH * 4);
    int*   slot1 = (int*)alloc(BATCH * 4);
    int*   slot2 = (int*)alloc(BATCH * 4);
    float* g1    = (float*)alloc(BATCH * 4);
    float* g2    = (float*)alloc(BATCH * 4);
    float* w1    = (float*)alloc(BATCH * 4);
    float* w2    = (float*)alloc(BATCH * 4);
    int*   hist1 = (int*)alloc(NGBLK * NEXP * 4);
    int*   hist2 = (int*)alloc(NGBLK * NEXP * 4);
    int*   off1  = (int*)alloc(NGBLK * NEXP * 4);
    int*   off2  = (int*)alloc(NGBLK * NEXP * 4);
    int*   T1    = (int*)alloc(NEXP * 4);
    bf16_t* h3   = h1;                       // h1 dead after GEMM1
    float* logits = (float*)dbuf;            // dbuf dead after expert GEMM

    const dim3 tb(32, 8);
    // weights: f32 [K][N] -> bf16 [Npad][K]
    transpose_cast_kernel<<<dim3(IN_DIM/32, HID/32, 1), tb, 0, stream>>>(W0, W0T, IN_DIM, HID, 0, 0);
    transpose_cast_kernel<<<dim3(HID/32, HID/32, 1),    tb, 0, stream>>>(W1, W1T, HID, HID, 0, 0);
    transpose_cast_kernel<<<dim3(HID/32, HID/32, NEXP), tb, 0, stream>>>(We, WeT, HID, HID,
                         (long long)HID * HID, (long long)HID * HID);
    transpose_cast_kernel<<<dim3(HID/32, NCPAD/32, 1),  tb, 0, stream>>>(Wp, WpT, HID, NC, 0, 0);

    // h1 = relu(x @ W0 + b0)         [8192,4096]x[4096,2048]
    gemm_wmma_bf16<true,  false, true,  true ><<<dim3(HID/128, BATCH/128, 1), 256, 0, stream>>>(
        x, W0T, h1, b0, BATCH, HID, IN_DIM, 0, 0, 0, 0);
    // h2 = relu(h1 @ W1 + b1)        [8192,2048]x[2048,2048]
    gemm_wmma_bf16<false, false, true,  true ><<<dim3(HID/128, BATCH/128, 1), 256, 0, stream>>>(
        h1, W1T, h2, b1, BATCH, HID, HID, 0, 0, 0, 0);

    // top-2 gating (deterministic, token-ordered capacity)
    gate_kernel<<<NGBLK, 256, 0, stream>>>(h2, Wg, idx1, idx2, g1, g2, hist1, hist2);
    scan_kernel<<<1, 1, 0, stream>>>(hist1, hist2, off1, off2, T1, NGBLK);
    assign_kernel<<<NGBLK, 1, 0, stream>>>(idx1, idx2, g1, g2, off1, off2, T1,
                                           slot1, slot2, w1, w2);
    dispatch_kernel<<<BATCH, 256, 0, stream>>>(h2, slot1, slot2, dbuf);

    // eo = buf @ We[e] + be[e]       batched [2816,2048]x[2048,2048] x6
    gemm_wmma_bf16<false, false, false, true ><<<dim3(HID/128, CAPP/128, NEXP), 256, 0, stream>>>(
        dbuf, WeT, eo, be, CAPP, HID, HID,
        (long long)CAPP * HID, (long long)HID * HID, (long long)CAPP * HID, HID);

    combine_kernel<<<BATCH, 256, 0, stream>>>(eo, slot1, slot2, w1, w2, h3);

    // logits = h3 @ Wp (bias folded into log_softmax)   [8192,2048]x[2048,1024]
    gemm_wmma_bf16<false, true,  false, false><<<dim3(NCPAD/128, BATCH/128, 1), 256, 0, stream>>>(
        h3, WpT, logits, nullptr, BATCH, NCPAD, HID, 0, 0, 0, 0);

    logsoftmax_kernel<<<BATCH, 256, 0, stream>>>(logits, bp, out);
}
